// WeightOnlyInt4Linear_41308995453016
// MI455X (gfx1250) — compile-verified, matrix operand order checked by reading, not run on hardware
//
#include <hip/hip_runtime.h>
#include <hip/hip_bf16.h>

#define IN_F   4096
#define OUT_F  11008
#define GS     128
#define NGRP   (IN_F / GS)   // 32
#define NK     (IN_F / 32)   // 128 K-iterations

typedef __attribute__((ext_vector_type(16))) _Float16 v16h;
typedef __attribute__((ext_vector_type(8)))  _Float16 v8h;
typedef __attribute__((ext_vector_type(2)))  _Float16 v2h;
typedef __attribute__((ext_vector_type(8)))  float    v8f;

#define BM 128
#define BN 128
#define BK 32
#define LDA 40                        // LDS row stride in halves (80 B, conflict-free)
#define A_HALVES   (BM * LDA)         // 5120 halves  (A tile region)
#define TILE_HALVES (A_HALVES + BN * LDA)  // 10240 halves per stage
#define STAGE_BYTES (TILE_HALVES * 2)      // 20480 B per stage

__global__ __launch_bounds__(256)
void w4a16_wmma_gemm(const _Float16* __restrict__ A,     // [M, IN_F] fp16
                     const uint32_t* __restrict__ W,     // [IN_F/8, OUT_F] packed int4
                     const _Float16* __restrict__ S,     // [OUT_F, NGRP]
                     const _Float16* __restrict__ Z,     // [OUT_F, NGRP]
                     const _Float16* __restrict__ Bias,  // [OUT_F]
                     _Float16* __restrict__ O)           // [M, OUT_F] fp16
{
    __shared__ __align__(16) _Float16 lds[2 * TILE_HALVES];   // 40 KB, double-buffered

    const int tid  = threadIdx.x;
    const int lane = tid & 31;
    const int wave = tid >> 5;
    const int wm   = wave & 1;        // 0..1 -> 64 M-rows each
    const int wn   = wave >> 1;       // 0..3 -> 32 N-cols each
    const int h    = lane >> 4;       // K half-select per WMMA 16-bit layout
    const int l15  = lane & 15;

    const int n0 = blockIdx.x * BN;
    const int m0 = blockIdx.y * BM;

    const int bn  = tid & 127;        // B-loader: column within N tile
    const int kwg = tid >> 7;         // B-loader: word rows {kwg, kwg+2}

    // ---- stage loaders --------------------------------------------------
    // A tile via CDNA5 async global->LDS (ASYNCcnt), 2 x 16B per thread.
    auto issueA = [&](int k0, int stage) {
        #pragma unroll
        for (int j = 0; j < 2; ++j) {
            int idx = tid + j * 256;          // 512 x 16B covers 128x32 fp16
            int row = idx >> 2;
            int q   = idx & 3;
            unsigned dst = (unsigned)(stage * STAGE_BYTES + (row * LDA + q * 8) * 2);
            unsigned long long src =
                (unsigned long long)(A + (size_t)(m0 + row) * IN_F + k0 + q * 8);
            asm volatile("global_load_async_to_lds_b128 %0, %1, off"
                         :: "v"(dst), "v"(src) : "memory");
        }
    };
    auto loadB = [&](int k0, uint32_t& w0, uint32_t& w1, float& s, float& z) {
        const int kw0 = k0 >> 3;
        const int g   = k0 >> 7;      // BK=32 never crosses GS=128
        s  = (float)S[(size_t)(n0 + bn) * NGRP + g];
        z  = (float)Z[(size_t)(n0 + bn) * NGRP + g];
        w0 = W[(size_t)(kw0 + kwg)     * OUT_F + n0 + bn];
        w1 = W[(size_t)(kw0 + kwg + 2) * OUT_F + n0 + bn];
    };
    // int4 -> fp16 via magic number: half(0x6400|q) == 1024+q, so
    // s*(q-z) = s*h - s*(1024+z); one v_pk_fma_f16 per nibble pair.
    auto storeB = [&](int stage, uint32_t w0, uint32_t w1, float s, float z) {
        const _Float16 sh = (_Float16)s;
        const _Float16 ch = (_Float16)(s * (1024.0f + z));
        #pragma unroll
        for (int r = 0; r < 2; ++r) {
            uint32_t w  = r ? w1 : w0;
            int      kr = kwg + r * 2;        // word row 0..3 -> klocal = kr*8
            v8h vals;
            #pragma unroll
            for (int j = 0; j < 4; ++j) {
                uint32_t t = w >> (8 * j);
                uint32_t p = (t & 0xFu) | ((t & 0xF0u) << 12) | 0x64006400u;
                v2h hv = __builtin_bit_cast(v2h, p);
                v2h rv = hv * sh - ch;
                vals[2 * j]     = rv[0];
                vals[2 * j + 1] = rv[1];
            }
            *(v8h*)&lds[stage * TILE_HALVES + A_HALVES + bn * LDA + kr * 8] = vals;
        }
    };

    // ---- accumulators ---------------------------------------------------
    v8f zero = {};
    v8f acc[4][2];
    #pragma unroll
    for (int mt = 0; mt < 4; ++mt)
        #pragma unroll
        for (int nt = 0; nt < 2; ++nt)
            acc[mt][nt] = zero;

    // ---- prologue: fill stage 0 -----------------------------------------
    issueA(0, 0);
    uint32_t w0, w1; float s, z;
    loadB(0, w0, w1, s, z);
    storeB(0, w0, w1, s, z);
    asm volatile("s_wait_asynccnt 0" ::: "memory");
    __syncthreads();

    // ---- main loop: 1 barrier / iter, double-buffered --------------------
    for (int i = 0; i < NK; ++i) {
        const int k0  = i * BK;
        const int cur = i & 1;
        const int nxt = cur ^ 1;
        const bool more = (i + 1) < NK;

        if (more) {
            issueA(k0 + BK, nxt);             // async A -> LDS[nxt], in flight
            loadB(k0 + BK, w0, w1, s, z);     // B words in flight during compute
            __builtin_prefetch(W + (size_t)((k0 >> 3) + 8 + kwg) * OUT_F + n0 + bn, 0, 1);
        }

        // -------- compute on LDS[cur] --------
        const _Float16* base = &lds[cur * TILE_HALVES];
        v16h af[4], bf[2];
        #pragma unroll
        for (int mt = 0; mt < 4; ++mt) {
            const _Float16* p = base + (wm * 64 + mt * 16 + l15) * LDA + h * 8;
            v8h lo = *(const v8h*)p;          // K = h*8 .. h*8+7
            v8h hi = *(const v8h*)(p + 16);   // K = 16+h*8 ..
            af[mt] = __builtin_shufflevector(lo, hi, 0,1,2,3,4,5,6,7,8,9,10,11,12,13,14,15);
        }
        #pragma unroll
        for (int nt = 0; nt < 2; ++nt) {
            const _Float16* p = base + A_HALVES + (wn * 32 + nt * 16 + l15) * LDA + h * 8;
            v8h lo = *(const v8h*)p;
            v8h hi = *(const v8h*)(p + 16);
            bf[nt] = __builtin_shufflevector(lo, hi, 0,1,2,3,4,5,6,7,8,9,10,11,12,13,14,15);
        }
        #pragma unroll
        for (int mt = 0; mt < 4; ++mt)
            #pragma unroll
            for (int nt = 0; nt < 2; ++nt)
                acc[mt][nt] = __builtin_amdgcn_wmma_f32_16x16x32_f16(
                    false, af[mt], false, bf[nt],
                    (short)0, acc[mt][nt], false, false);

        // -------- stage i+1 into LDS[nxt] --------
        if (more) {
            storeB(nxt, w0, w1, s, z);
            asm volatile("s_wait_asynccnt 0" ::: "memory");
        }
        __syncthreads();
    }

    // ---- epilogue: bias add, fp16 store (C/D: VGPR j -> M=j+8h, lane&15 -> N)
    #pragma unroll
    for (int nt = 0; nt < 2; ++nt) {
        int   col = n0 + wn * 32 + nt * 16 + l15;
        float b   = (float)Bias[col];
        #pragma unroll
        for (int mt = 0; mt < 4; ++mt) {
            int rbase = m0 + wm * 64 + mt * 16 + h * 8;
            #pragma unroll
            for (int j = 0; j < 8; ++j)
                O[(size_t)(rbase + j) * OUT_F + col] = (_Float16)(acc[mt][nt][j] + b);
        }
    }
}

extern "C" void kernel_launch(void* const* d_in, const int* in_sizes, int n_in,
                              void* d_out, int out_size, void* d_ws, size_t ws_size,
                              hipStream_t stream) {
    const _Float16* A    = (const _Float16*)d_in[0];
    const uint32_t* W    = (const uint32_t*)d_in[1];
    const _Float16* S    = (const _Float16*)d_in[2];
    const _Float16* Z    = (const _Float16*)d_in[3];
    const _Float16* Bias = (const _Float16*)d_in[4];
    _Float16*       O    = (_Float16*)d_out;

    const int M = in_sizes[0] / IN_F;         // 8192 tokens
    dim3 grid(OUT_F / BN, M / BM);            // 86 x 64 workgroups
    w4a16_wmma_gemm<<<grid, 256, 0, stream>>>(A, W, S, Z, Bias, O);
}
